// GCN_80023830659293
// MI455X (gfx1250) — matrix operand order repaired and checked by reading, not compiled, not run on hardware
//
#include <hip/hip_runtime.h>

// ---------------------------------------------------------------------------
// Types for CDNA5 WMMA (wave32)
// ---------------------------------------------------------------------------
typedef __attribute__((ext_vector_type(16))) __bf16 v16bf;
typedef __attribute__((ext_vector_type(8)))  float  v8f;

typedef unsigned int   u32;
typedef unsigned short u16;

#define NN   50000
#define EE   200000
#define FIN  64
#define EDIM 16
#define DD   256
#define HIDD 64
#define GG   2000

// ---------------------------------------------------------------------------
// Helpers
// ---------------------------------------------------------------------------
__device__ __forceinline__ u16 f2bf(float f) {
    u32 b = __float_as_uint(f);
    u32 r = b + 0x7FFFu + ((b >> 16) & 1u);   // round-to-nearest-even
    return (u16)(r >> 16);
}
__device__ __forceinline__ float bf2f(u16 u) {
    return __uint_as_float(((u32)u) << 16);
}
// order-preserving float <-> uint encoding for atomicMax on floats
__device__ __forceinline__ u32 fenc(float f) {
    u32 b = __float_as_uint(f);
    return (b & 0x80000000u) ? ~b : (b | 0x80000000u);
}
__device__ __forceinline__ float fdec(u32 u) {
    return __uint_as_float((u & 0x80000000u) ? (u ^ 0x80000000u) : ~u);
}
#define ENC_NEG_INF 0x007FFFFFu   // fenc(-inf)

// ---------------------------------------------------------------------------
// Weight repack: W[K x Nout] (f32) -> fragment-ordered bf16:
//   Wp[((ntile*ktiles + kt)*32 + lane)*16 + j] = W[kt*32 + (lane>>4)*16 + j][ntile*16 + (lane&15)]
// so each lane's 16-element B fragment is 32 contiguous bytes (2 x b128 loads).
// Zero-pads K up to ktiles*32.
// ---------------------------------------------------------------------------
__global__ void repack_w_k(const float* __restrict__ W, u16* __restrict__ Wp,
                           int K, int Nout, int ktiles, int total)
{
    int i = blockIdx.x * blockDim.x + threadIdx.x;
    if (i >= total) return;
    int j    = i & 15;
    int lane = (i >> 4) & 31;
    int kt   = (i >> 9) % ktiles;
    int nt   = (i >> 9) / ktiles;
    int k = kt * 32 + (lane >> 4) * 16 + j;
    int n = nt * 16 + (lane & 15);
    Wp[i] = (k < K) ? f2bf(W[(size_t)k * Nout + n]) : (u16)0;
}

// ---------------------------------------------------------------------------
// WMMA GEMM:  Out[M x Nout] = act( A[M x LDA](f32, K zero-padded to KP) @ W + bias )
// Block: 256 threads = 8 waves; covers 32 rows x 128 cols.
// Each wave: 2 M-tiles x 1 N-tile, sharing one B fragment per K step.
// KP, LDA compile-time -> fully unrolled K loop, no bounds checks.
// M need not be a multiple of 32 (tail clamped/guarded wave-uniformly); M%16==0.
// ---------------------------------------------------------------------------
template <int KP, int LDA, bool OUTBF, bool RELU>
__global__ __launch_bounds__(256) void wmma_gemm(
    const float* __restrict__ A, const uint4* __restrict__ Wp,
    const float* __restrict__ bias, void* __restrict__ Out,
    int M, int Nout)
{
    __shared__ u32 lds_a[32][20];   // 32 rows x 16 u32 (32 bf16) + 4 u32 pad (80B stride)

    const int tid  = threadIdx.x;
    const int wid  = tid >> 5;
    const int lane = tid & 31;
    const int m0   = blockIdx.x * 32;
    const int ntile = blockIdx.y * 8 + wid;
    const int n0   = ntile * 16;
    const bool activeN = (n0 < Nout);           // wave-uniform
    const bool tile1ok = (m0 + 16 < M);         // wave-uniform (M % 16 == 0)

    const int hh   = lane >> 4;
    const int mrow = lane & 15;
    const int col  = n0 + (lane & 15);
    constexpr int KT = KP / 32;

    v8f acc0 = {}, acc1 = {};

    #pragma unroll
    for (int kt = 0; kt < KT; ++kt) {
        const int k0 = kt * 32;
        // ---- stage 32x32 A chunk into LDS (f32 -> packed bf16x2), coalesced ----
        #pragma unroll
        for (int t = tid; t < 512; t += 256) {
            int r = t >> 4, cp = t & 15;
            int cc = k0 + cp * 2;
            int rm = m0 + r; if (rm >= M) rm = M - 1;   // clamp; tail rows discarded at store
            float x0 = 0.0f, x1 = 0.0f;
            if (cc + 1 < LDA + 1) {                      // cc < LDA (LDA even)
                if (cc < LDA) {
                    float2 xy = *(const float2*)(A + (size_t)rm * LDA + cc);
                    x0 = xy.x; x1 = xy.y;
                }
            }
            lds_a[r][cp] = (u32)f2bf(x0) | ((u32)f2bf(x1) << 16);
            if (kt + 1 < KT)
                __builtin_prefetch(A + (size_t)rm * LDA + cc + 32, 0, 1);
        }
        __syncthreads();

        if (activeN) {
            // A fragments: lane (hh,mrow) holds K = {hh*16..hh*16+15}
            union { v16bf v; uint4 q[2]; } a0, a1, bfr;
            a0.q[0] = *(const uint4*)&lds_a[mrow][hh * 8];
            a0.q[1] = *(const uint4*)&lds_a[mrow][hh * 8 + 4];
            a1.q[0] = *(const uint4*)&lds_a[16 + mrow][hh * 8];
            a1.q[1] = *(const uint4*)&lds_a[16 + mrow][hh * 8 + 4];
            // B fragment: 32 contiguous bytes per lane, coalesced b128 x2
            size_t bi = ((size_t)((size_t)ntile * KT + kt) * 32 + lane) * 2;
            bfr.q[0] = Wp[bi];
            bfr.q[1] = Wp[bi + 1];
            acc0 = __builtin_amdgcn_wmma_f32_16x16x32_bf16(
                false, a0.v, false, bfr.v, (short)0, acc0, false, false);
            acc1 = __builtin_amdgcn_wmma_f32_16x16x32_bf16(
                false, a1.v, false, bfr.v, (short)0, acc1, false, false);
        }
        __syncthreads();
    }

    if (activeN) {
        float bv = bias ? bias[col] : 0.0f;
        #pragma unroll
        for (int r = 0; r < 8; ++r) {       // VGPR r: lanes0-15 -> M=r, lanes16-31 -> M=r+8
            int row = m0 + r + hh * 8;
            float val = acc0[r] + bv;
            if (RELU) val = fmaxf(val, 0.0f);
            if (OUTBF) ((u16*)Out)[(size_t)row * Nout + col] = f2bf(val);
            else       ((float*)Out)[(size_t)row * Nout + col] = val;
        }
        if (tile1ok) {
            #pragma unroll
            for (int r = 0; r < 8; ++r) {
                int row = m0 + 16 + r + hh * 8;
                float val = acc1[r] + bv;
                if (RELU) val = fmaxf(val, 0.0f);
                if (OUTBF) ((u16*)Out)[(size_t)row * Nout + col] = f2bf(val);
                else       ((float*)Out)[(size_t)row * Nout + col] = val;
            }
        }
    }
}

// ---------------------------------------------------------------------------
// Attention kernels (wave-per-edge; 32 lanes x 8 features = 256 = D; head = lane>>3)
// ---------------------------------------------------------------------------
__global__ __launch_bounds__(256) void attn_alpha(
    const float* __restrict__ q, const float* __restrict__ k, const u16* __restrict__ e,
    const int* __restrict__ src, const int* __restrict__ dst,
    float* __restrict__ alpha, u32* __restrict__ amax, int E_)
{
    int wid = threadIdx.x >> 5, lane = threadIdx.x & 31;
    int eid = blockIdx.x * 8 + wid;
    if (eid >= E_) return;
    int s = src[eid], d = dst[eid];
    int f0 = lane * 8;
    const float4* qp = (const float4*)(q + (size_t)d * DD + f0);
    const float4* kp = (const float4*)(k + (size_t)s * DD + f0);
    const uint4*  ep = (const uint4*)(e + (size_t)eid * DD + f0);
    float4 q0 = qp[0], q1 = qp[1];
    float4 k0 = kp[0], k1 = kp[1];
    uint4  eu = ep[0];
    float p =
        q0.x * (k0.x + bf2f((u16)(eu.x & 0xFFFF))) +
        q0.y * (k0.y + bf2f((u16)(eu.x >> 16)))    +
        q0.z * (k0.z + bf2f((u16)(eu.y & 0xFFFF))) +
        q0.w * (k0.w + bf2f((u16)(eu.y >> 16)))    +
        q1.x * (k1.x + bf2f((u16)(eu.z & 0xFFFF))) +
        q1.y * (k1.y + bf2f((u16)(eu.z >> 16)))    +
        q1.z * (k1.z + bf2f((u16)(eu.w & 0xFFFF))) +
        q1.w * (k1.w + bf2f((u16)(eu.w >> 16)));
    p += __shfl_xor(p, 4, 32);
    p += __shfl_xor(p, 2, 32);
    p += __shfl_xor(p, 1, 32);
    if ((lane & 7) == 0) {
        int h = lane >> 3;
        float a = p * 0.125f;                 // 1/sqrt(64)
        alpha[(size_t)eid * 4 + h] = a;
        atomicMax(&amax[(size_t)d * 4 + h], fenc(a));
    }
}

__global__ void attn_exp(
    float* __restrict__ al, const u32* __restrict__ amax,
    float* __restrict__ asum, const int* __restrict__ dst, int E_)
{
    int i = blockIdx.x * blockDim.x + threadIdx.x;
    if (i >= E_ * 4) return;
    int eid = i >> 2, h = i & 3;
    int d = dst[eid];
    float m  = fdec(amax[(size_t)d * 4 + h]);
    float ex = __expf(al[i] - m);
    al[i] = ex;                               // alpha buffer becomes aexp in place
    atomicAdd(&asum[(size_t)d * 4 + h], ex);
}

__global__ __launch_bounds__(256) void attn_scatter(
    const float* __restrict__ v, const u16* __restrict__ e,
    const float* __restrict__ aexp, const float* __restrict__ asum,
    const int* __restrict__ src, const int* __restrict__ dst,
    float* __restrict__ out, int E_)
{
    int wid = threadIdx.x >> 5, lane = threadIdx.x & 31;
    int eid = blockIdx.x * 8 + wid;
    if (eid >= E_) return;
    int s = src[eid], d = dst[eid];
    int h = lane >> 3;
    float w = aexp[(size_t)eid * 4 + h] / asum[(size_t)d * 4 + h];
    int f0 = lane * 8;
    const float4* vp = (const float4*)(v + (size_t)s * DD + f0);
    const uint4*  ep = (const uint4*)(e + (size_t)eid * DD + f0);
    float4 v0 = vp[0], v1 = vp[1];
    uint4  eu = ep[0];
    float* op = out + (size_t)d * DD + f0;
    atomicAdd(&op[0], (v0.x + bf2f((u16)(eu.x & 0xFFFF))) * w);
    atomicAdd(&op[1], (v0.y + bf2f((u16)(eu.x >> 16)))    * w);
    atomicAdd(&op[2], (v0.z + bf2f((u16)(eu.y & 0xFFFF))) * w);
    atomicAdd(&op[3], (v0.w + bf2f((u16)(eu.y >> 16)))    * w);
    atomicAdd(&op[4], (v1.x + bf2f((u16)(eu.z & 0xFFFF))) * w);
    atomicAdd(&op[5], (v1.y + bf2f((u16)(eu.z >> 16)))    * w);
    atomicAdd(&op[6], (v1.z + bf2f((u16)(eu.w & 0xFFFF))) * w);
    atomicAdd(&op[7], (v1.w + bf2f((u16)(eu.w >> 16)))    * w);
}

// ---------------------------------------------------------------------------
// Misc elementwise kernels
// ---------------------------------------------------------------------------
__global__ void fill_u32_k(u32* __restrict__ p, u32 v, int n) {
    int i = blockIdx.x * blockDim.x + threadIdx.x;
    if (i < n) p[i] = v;
}
__global__ void tanh4_k(float4* __restrict__ h, int n4) {
    int i = blockIdx.x * blockDim.x + threadIdx.x;
    if (i >= n4) return;
    float4 t = h[i];
    t.x = tanhf(t.x); t.y = tanhf(t.y); t.z = tanhf(t.z); t.w = tanhf(t.w);
    h[i] = t;
}
__global__ void pool_acc_k(const float* __restrict__ h, const int* __restrict__ batch,
                           float* __restrict__ pooled, int total) {
    int i = blockIdx.x * blockDim.x + threadIdx.x;
    if (i >= total) return;
    int n = i >> 8;
    atomicAdd(&pooled[(size_t)batch[n] * DD + (i & 255)], h[i]);
}
__global__ void pool_cnt_k(const int* __restrict__ batch, float* __restrict__ cnt, int n) {
    int i = blockIdx.x * blockDim.x + threadIdx.x;
    if (i < n) atomicAdd(&cnt[batch[i]], 1.0f);
}
__global__ void pool_div_k(float* __restrict__ pooled, const float* __restrict__ cnt, int total) {
    int i = blockIdx.x * blockDim.x + threadIdx.x;
    if (i >= total) return;
    pooled[i] /= fmaxf(cnt[i >> 8], 1.0f);
}
__global__ void mlp2_k(const float* __restrict__ hid, const float* __restrict__ W2,
                       const float* __restrict__ b2, float* __restrict__ out, int G_) {
    int i = blockIdx.x * blockDim.x + threadIdx.x;
    if (i >= G_ * 2) return;
    int g = i >> 1, c = i & 1;
    float sum = b2[c];
    const float* hp = hid + (size_t)g * HIDD;
    #pragma unroll 8
    for (int j = 0; j < HIDD; ++j) sum += hp[j] * W2[j * 2 + c];
    out[i] = sum;
}

// ---------------------------------------------------------------------------
// Host driver
// ---------------------------------------------------------------------------
extern "C" void kernel_launch(void* const* d_in, const int* in_sizes, int n_in,
                              void* d_out, int out_size, void* d_ws, size_t ws_size,
                              hipStream_t stream)
{
    (void)in_sizes; (void)n_in; (void)out_size; (void)ws_size;

    const float* x         = (const float*)d_in[0];
    const int*   edge_idx  = (const int*)d_in[1];
    const float* edge_attr = (const float*)d_in[2];
    const int*   batch     = (const int*)d_in[3];
    const int*   srcp = edge_idx;
    const int*   dstp = edge_idx + EE;

    // workspace sub-allocation
    size_t off = 0;
    auto alloc = [&](size_t bytes) -> void* {
        void* p = (char*)d_ws + off;
        off = (off + bytes + 255) & ~(size_t)255;
        return p;
    };
    float* qb    = (float*)alloc((size_t)NN * DD * 4);
    float* kb    = (float*)alloc((size_t)NN * DD * 4);
    float* vb    = (float*)alloc((size_t)NN * DD * 4);
    float* hA    = (float*)alloc((size_t)NN * DD * 4);
    float* hB    = (float*)alloc((size_t)NN * DD * 4);
    u16*   ebuf  = (u16*)  alloc((size_t)EE * DD * 2);
    float* aexpb = (float*)alloc((size_t)EE * 4 * 4);
    u32*   amaxb = (u32*)  alloc((size_t)NN * 4 * 4);
    float* asumb = (float*)alloc((size_t)NN * 4 * 4);
    float* pooled= (float*)alloc((size_t)GG * DD * 4);
    float* cntb  = (float*)alloc((size_t)GG * 4);
    float* hidb  = (float*)alloc((size_t)GG * HIDD * 4);

    // repack all GEMM weights into fragment-ordered bf16 (zero-padded K)
    auto repack = [&](const void* W, int K, int Nout) -> const uint4* {
        int ktiles = (K + 31) / 32;
        int total  = (Nout / 16) * ktiles * 512;          // = Nout * KP
        u16* dst = (u16*)alloc((size_t)total * 2);
        repack_w_k<<<(total + 255) / 256, 256, 0, stream>>>(
            (const float*)W, dst, K, Nout, ktiles, total);
        return (const uint4*)dst;
    };
    const uint4 *WqB[3], *WkB[3], *WvB[3], *WeB[3], *WsB[3];
    const float *bq[3], *bk[3], *bv[3], *bsk[3];
    const int fin[3] = { FIN, DD, DD };
    for (int l = 0; l < 3; ++l) {
        int base = 4 + 9 * l;
        WqB[l] = repack(d_in[base + 0], fin[l], DD);
        bq[l]  = (const float*)d_in[base + 1];
        WkB[l] = repack(d_in[base + 2], fin[l], DD);
        bk[l]  = (const float*)d_in[base + 3];
        WvB[l] = repack(d_in[base + 4], fin[l], DD);
        bv[l]  = (const float*)d_in[base + 5];
        WeB[l] = repack(d_in[base + 6], EDIM, DD);
        WsB[l] = repack(d_in[base + 7], fin[l], DD);
        bsk[l] = (const float*)d_in[base + 8];
    }
    const uint4* W1B = repack(d_in[31], DD, HIDD);
    const float* b1  = (const float*)d_in[32];
    const float* W2  = (const float*)d_in[33];
    const float* b2  = (const float*)d_in[34];

    const dim3 gridN((NN + 31) / 32, 2);   // node GEMMs: Nout=256
    const dim3 gridE((EE + 31) / 32, 2);   // edge GEMM
    const dim3 gridP((GG + 31) / 32, 1);   // MLP1: Nout=64

    // ---- 3 TransformerConv layers ----
    const float* hin = x;
    float* hnext = hA;
    float* hother = hB;
    for (int l = 0; l < 3; ++l) {
        if (l == 0) {
            wmma_gemm<64, 64, false, false><<<gridN, 256, 0, stream>>>(hin, WqB[l], bq[l],  qb,    NN, DD);
            wmma_gemm<64, 64, false, false><<<gridN, 256, 0, stream>>>(hin, WkB[l], bk[l],  kb,    NN, DD);
            wmma_gemm<64, 64, false, false><<<gridN, 256, 0, stream>>>(hin, WvB[l], bv[l],  vb,    NN, DD);
            wmma_gemm<64, 64, false, false><<<gridN, 256, 0, stream>>>(hin, WsB[l], bsk[l], hnext, NN, DD);
        } else {
            wmma_gemm<256, 256, false, false><<<gridN, 256, 0, stream>>>(hin, WqB[l], bq[l],  qb,    NN, DD);
            wmma_gemm<256, 256, false, false><<<gridN, 256, 0, stream>>>(hin, WkB[l], bk[l],  kb,    NN, DD);
            wmma_gemm<256, 256, false, false><<<gridN, 256, 0, stream>>>(hin, WvB[l], bv[l],  vb,    NN, DD);
            wmma_gemm<256, 256, false, false><<<gridN, 256, 0, stream>>>(hin, WsB[l], bsk[l], hnext, NN, DD);
        }
        wmma_gemm<32, 16, true, false><<<gridE, 256, 0, stream>>>(edge_attr, WeB[l], nullptr, ebuf, EE, DD);

        fill_u32_k<<<(NN * 4 + 255) / 256, 256, 0, stream>>>(amaxb, ENC_NEG_INF, NN * 4);
        fill_u32_k<<<(NN * 4 + 255) / 256, 256, 0, stream>>>((u32*)asumb, 0u, NN * 4);

        attn_alpha<<<EE / 8, 256, 0, stream>>>(qb, kb, ebuf, srcp, dstp, aexpb, amaxb, EE);
        attn_exp<<<(EE * 4 + 255) / 256, 256, 0, stream>>>(aexpb, amaxb, asumb, dstp, EE);
        attn_scatter<<<EE / 8, 256, 0, stream>>>(vb, ebuf, aexpb, asumb, srcp, dstp, hnext, EE);

        if (l < 2) tanh4_k<<<(NN * DD / 4 + 255) / 256, 256, 0, stream>>>((float4*)hnext, NN * DD / 4);

        hin = hnext;
        float* t = hnext; hnext = hother; hother = t;
    }
    const float* hfin = hin;

    // ---- mean pool over batch ----
    fill_u32_k<<<(GG * DD + 255) / 256, 256, 0, stream>>>((u32*)pooled, 0u, GG * DD);
    fill_u32_k<<<(GG + 255) / 256, 256, 0, stream>>>((u32*)cntb, 0u, GG);
    pool_acc_k<<<(NN * DD + 255) / 256, 256, 0, stream>>>(hfin, batch, pooled, NN * DD);
    pool_cnt_k<<<(NN + 255) / 256, 256, 0, stream>>>(batch, cntb, NN);
    pool_div_k<<<(GG * DD + 255) / 256, 256, 0, stream>>>(pooled, cntb, GG * DD);

    // ---- MLP head ----
    wmma_gemm<256, 256, false, true><<<gridP, 256, 0, stream>>>(pooled, W1B, b1, hidb, GG, HIDD);
    mlp2_k<<<(GG * 2 + 255) / 256, 256, 0, stream>>>(hidb, W2, b2, (float*)d_out, GG);
}
